// InfoNCELoss_15375982920193
// MI455X (gfx1250) — compile-verified, matrix operand order checked by reading, not run on hardware
//
#include <hip/hip_runtime.h>

// InfoNCE loss, fused flash-style streaming over the Gram matrix.
// Precision: fp32 -> (bf16 hi, bf16 lo) split; S = hi*hi^T + hi*lo^T + lo*hi^T
// via v_wmma_f32_16x16x32_bf16 (error ~2^-16 relative, fp32-class).
// Data movement: gfx1250 GLOBAL_LOAD_ASYNC_TO_LDS_B128 (ASYNCcnt-tracked,
// zero data VGPRs) with a double-buffered LDS B-block; one wait+barrier/iter.
// Workspace: 2*B*D*2 bytes (hi/lo) + 3*8*B*4 bytes (partials) ~= 17.6 MB.

typedef __attribute__((ext_vector_type(16))) __bf16 v16bf;
typedef __attribute__((ext_vector_type(8)))  float  v8f;

constexpr int   kB       = 8192;
constexpr int   kD       = 512;
constexpr float kInvT    = 1.0f / 0.07f;        // 14.2857...
constexpr float kLMax    = 1.0f / 0.07f;        // max logit (normalized rows)
constexpr int   kTimeThr = 5000000;

constexpr int ROWS        = 64;                  // rows per workgroup (4 row-tiles)
constexpr int CTILE       = 32;                  // columns per iteration (2 col-tiles)
constexpr int COL_SPLIT   = 4;                   // column-range split for occupancy
constexpr int COLS_PER_WG = kB / COL_SPLIT;      // 2048
constexpr int ITERS       = COLS_PER_WG / CTILE; // 64
constexpr int STRIDE      = kD + 8;              // 520 halves; pad dodges LDS bank conflicts
constexpr int SLOTS       = COL_SPLIT * 2;       // deterministic partial slots per row
constexpr int BCHUNKS     = (CTILE * (kD / 8)) / 256;  // 16B chunks per thread per type = 8
constexpr int BBUF_BYTES  = 2 * CTILE * STRIDE * 2;    // one B double-buffer slot (hi+lo)
constexpr int BSTEP_BYTES = CTILE * kD * 2;            // global stride between B blocks

// ---------------------------------------------------------------- split

// fp32 -> bf16 hi/lo split (round-to-nearest-even, manual bit twiddle)
__global__ void infonce_split_kernel(const float* __restrict__ x,
                                     unsigned short* __restrict__ hi,
                                     unsigned short* __restrict__ lo, int n) {
  int i = blockIdx.x * blockDim.x + threadIdx.x;
  if (i >= n) return;
  float v = x[i];
  unsigned u = __float_as_uint(v);
  unsigned h = (u + 0x7FFFu + ((u >> 16) & 1u)) >> 16;
  float hf = __uint_as_float(h << 16);
  float r = v - hf;
  unsigned ur = __float_as_uint(r);
  unsigned l = (ur + 0x7FFFu + ((ur >> 16) & 1u)) >> 16;
  hi[i] = (unsigned short)h;
  lo[i] = (unsigned short)l;
}

// ---------------------------------------------------------------- helpers

// gfx1250 async global->LDS copy, GVS addressing (SGPR base + VGPR byte offset).
// vdst = wave-relative LDS byte address; tracked by ASYNCcnt, no data VGPRs.
static __device__ inline void async_load_b128(unsigned lds_addr, unsigned voff,
                                              const void* sbase) {
  asm volatile("global_load_async_to_lds_b128 %0, %1, %2"
               :: "v"(lds_addr), "v"(voff), "s"(sbase)
               : "memory");
}
static __device__ inline void wait_async0() {
  asm volatile("s_wait_asynccnt 0" ::: "memory");
}

// Load a 16x32 bf16 fragment (A-operand pattern) from padded LDS.
// Lane layout per ISA: lanes 0-15 carry K=[k0..k0+7]+[k0+16..k0+23],
// lanes 16-31 carry K=[k0+8..k0+15]+[k0+24..k0+31]; row = lane%16 (+tile base).
static __device__ inline v16bf load_frag(const unsigned short* __restrict__ base,
                                         int row, int k) {
  union { uint4 q[2]; v16bf v; } u;
  const unsigned short* p = base + row * STRIDE + k;
  u.q[0] = *reinterpret_cast<const uint4*>(p);
  u.q[1] = *reinterpret_cast<const uint4*>(p + 16);
  return u.v;
}

static __device__ inline v8f wmma_bf16(v16bf a, v16bf b, v8f c) {
  return __builtin_amdgcn_wmma_f32_16x16x32_bf16(
      /*neg_a=*/false, a, /*neg_b=*/false, b,
      /*c_mod=*/(short)0, c, /*reuse_a=*/false, /*reuse_b=*/false);
}

// ---------------------------------------------------------------- main kernel

__global__ __launch_bounds__(256, 1)
void infonce_gram_kernel(const unsigned short* __restrict__ ghi,
                         const unsigned short* __restrict__ glo,
                         const int* __restrict__ seq,
                         const int* __restrict__ tmid,
                         float* __restrict__ Zp,  // [SLOTS][kB]
                         float* __restrict__ Pp,  // [SLOTS][kB]
                         float* __restrict__ Np)  // [SLOTS][kB]
{
  // LDS: A-block (64 rows, hi+lo) resident; B-block (32 cols, hi+lo) double-buffered.
  __shared__ __align__(16) unsigned short smem[(ROWS * 2 + CTILE * 4) * STRIDE];
  unsigned short* sAhi = smem;
  unsigned short* sAlo = smem + ROWS * STRIDE;
  unsigned short* sB   = smem + 2 * ROWS * STRIDE; // [buf][type][CTILE][STRIDE]

  const int tid  = threadIdx.x;
  const int wid  = tid >> 5;
  const int lane = tid & 31;
  const int rt   = wid >> 1;        // row-tile 0..3
  const int cw   = wid & 1;         // col-tile within the 32-wide stage
  const int rb   = (int)blockIdx.x / COL_SPLIT;
  const int cs   = (int)blockIdx.x % COL_SPLIT;
  const int row0 = rb * ROWS;
  const int colBase = cs * COLS_PER_WG;
  const int slot = cs * 2 + cw;

  // Wave-relative LDS byte addresses: low 32 bits of the generic pointer to
  // LDS are the LDS offset (aperture lives in bits [63:32] per ISA 10.2).
  const unsigned ldsBase  = (unsigned)(uintptr_t)(void*)smem;
  const unsigned ldsAlo   = ldsBase + (unsigned)(ROWS * STRIDE * 2);
  const unsigned ldsBBase = ldsBase + (unsigned)(2 * ROWS * STRIDE * 2);
  const unsigned ldsBtype = (unsigned)(CTILE * STRIDE * 2);

  // per-thread B staging geometry (16B chunks): id = t*256+tid -> (col, kchunk)
  unsigned bDstB[BCHUNKS];  // LDS byte offset within one [CTILE][STRIDE] region
  unsigned bVoff[BCHUNKS];  // global byte offset of the NEXT block to issue
#pragma unroll
  for (int t = 0; t < BCHUNKS; ++t) {
    int id  = t * 256 + tid;
    int col = id >> 6;
    int kc  = id & 63;
    bDstB[t] = (unsigned)((col * STRIDE + kc * 8) * 2);
    bVoff[t] = (unsigned)(((colBase + col) * (kD / 8) + kc) * 16);
  }

  // ---- prologue: async-stage resident A-block (32 issues) + B block it=0 (16)
#pragma unroll
  for (int t = 0; t < (ROWS * 64) / 256; ++t) {
    int id  = t * 256 + tid;
    int r   = id >> 6;        // row within block
    int kc  = id & 63;        // 16B chunk within row
    unsigned dst  = (unsigned)((r * STRIDE + kc * 8) * 2);
    unsigned voff = (unsigned)(((row0 + r) * (kD / 8) + kc) * 16);
    async_load_b128(ldsBase + dst, voff, ghi);
    async_load_b128(ldsAlo  + dst, voff, glo);
  }
#pragma unroll
  for (int t = 0; t < BCHUNKS; ++t) {
    async_load_b128(ldsBBase + bDstB[t],            bVoff[t], ghi);
    async_load_b128(ldsBBase + ldsBtype + bDstB[t], bVoff[t], glo);
    bVoff[t] += (unsigned)BSTEP_BYTES;   // now points at block it=1
  }

  // per-lane C ownership: vgpr r -> (m = r + 8*(lane>=16), n = lane&15)
  const int giBase = row0 + rt * 16 + ((lane >> 4) << 3); // gi = giBase + r
  int   si[8], ti[8];
#pragma unroll
  for (int r = 0; r < 8; ++r) { si[r] = seq[giBase + r]; ti[r] = tmid[giBase + r]; }

  float zr[8], pr[8], nr[8];
#pragma unroll
  for (int r = 0; r < 8; ++r) { zr[r] = 0.f; pr[r] = 0.f; nr[r] = 0.f; }

  const int rowA  = rt * 16 + (lane & 15);
  const int rowB  = cw * 16 + (lane & 15);
  const int halfk = (lane >> 4) * 8;

  for (int it = 0; it < ITERS; ++it) {
    wait_async0();     // this wave's async writes into buf[it&1] have landed
    __syncthreads();   // => all waves' writes landed; everyone done reading buf[(it+1)&1]

    // ---- issue async loads for it+1 into the other buffer (hidden by compute)
    if (it + 1 < ITERS) {
      const unsigned bufOff = ldsBBase + (unsigned)(((it + 1) & 1) * BBUF_BYTES);
#pragma unroll
      for (int t = 0; t < BCHUNKS; ++t) {
        async_load_b128(bufOff + bDstB[t],            bVoff[t], ghi);
        async_load_b128(bufOff + ldsBtype + bDstB[t], bVoff[t], glo);
        bVoff[t] += (unsigned)BSTEP_BYTES;
      }
    }

    const int gj = colBase + it * CTILE + cw * 16 + (lane & 15);
    const int sj = seq[gj];
    const int tj = tmid[gj];

    const unsigned short* dHi = sB + (it & 1) * (2 * CTILE * STRIDE);
    const unsigned short* dLo = dHi + CTILE * STRIDE;

    v8f c = {0.f, 0.f, 0.f, 0.f, 0.f, 0.f, 0.f, 0.f};
#pragma unroll
    for (int ks = 0; ks < kD / 32; ++ks) {
      const int k = ks * 32 + halfk;
      v16bf ah = load_frag(sAhi, rowA, k);
      v16bf al = load_frag(sAlo, rowA, k);
      v16bf bh = load_frag(dHi, rowB, k);
      v16bf bl = load_frag(dLo, rowB, k);
      c = wmma_bf16(ah, bh, c);
      c = wmma_bf16(ah, bl, c);
      c = wmma_bf16(al, bh, c);
    }

    // fused mask + streaming logsumexp / positive-pair accumulation
#pragma unroll
    for (int r = 0; r < 8; ++r) {
      float l   = c[r] * kInvT;
      int   gi  = giBase + r;
      bool  diag = (gi == gj);
      float e = __expf(l - kLMax);           // l - LMax in [-2/T, 0]: safe
      zr[r] += diag ? 0.f : e;
      int dt = ti[r] - tj; dt = dt < 0 ? -dt : dt;
      bool pos = (!diag) && (si[r] == sj) && (dt < kTimeThr);
      pr[r] += pos ? l : 0.f;
      nr[r] += pos ? 1.f : 0.f;
    }
  }

  // reduce across the 16 lanes sharing a row (xor masks stay within each half)
#pragma unroll
  for (int r = 0; r < 8; ++r) {
    float z = zr[r], p = pr[r], n = nr[r];
#pragma unroll
    for (int m = 8; m >= 1; m >>= 1) {
      z += __shfl_xor(z, m, 32);
      p += __shfl_xor(p, m, 32);
      n += __shfl_xor(n, m, 32);
    }
    if ((lane & 15) == 0) {
      int gi = giBase + r;           // lane 0 -> row r, lane 16 -> row r+8
      Zp[slot * kB + gi] = z;        // every (slot,gi) written exactly once:
      Pp[slot * kB + gi] = p;        // no pre-zeroing needed, fully deterministic
      Np[slot * kB + gi] = n;
    }
  }
}

// ---------------------------------------------------------------- finalize

__global__ __launch_bounds__(256)
void infonce_finalize_kernel(const float* __restrict__ Zp,
                             const float* __restrict__ Pp,
                             const float* __restrict__ Np,
                             float* __restrict__ out) {
  __shared__ double snum[256];
  __shared__ double sden[256];
  int tid = threadIdx.x;
  double num = 0.0, den = 0.0;
  for (int i = tid; i < kB; i += 256) {
    float Z = 0.f, P = 0.f, N = 0.f;
#pragma unroll
    for (int s = 0; s < SLOTS; ++s) {
      Z += Zp[s * kB + i];
      P += Pp[s * kB + i];
      N += Np[s * kB + i];
    }
    num += (double)N * (log((double)Z) + (double)kLMax) - (double)P;
    den += (double)N;
  }
  snum[tid] = num; sden[tid] = den;
  __syncthreads();
  for (int s = 128; s > 0; s >>= 1) {
    if (tid < s) { snum[tid] += snum[tid + s]; sden[tid] += sden[tid + s]; }
    __syncthreads();
  }
  if (tid == 0) {
    double d = sden[0] >= 1.0 ? sden[0] : 1.0;
    out[0] = (float)(snum[0] / d);
  }
}

// ---------------------------------------------------------------- launch

extern "C" void kernel_launch(void* const* d_in, const int* in_sizes, int n_in,
                              void* d_out, int out_size, void* d_ws, size_t ws_size,
                              hipStream_t stream) {
  const float* desc = (const float*)d_in[0];
  const int*   seq  = (const int*)d_in[1];
  const int*   tmid = (const int*)d_in[2];
  float* out = (float*)d_out;

  char* ws = (char*)d_ws;
  unsigned short* hi = (unsigned short*)ws;                       //  8 MB
  unsigned short* lo = hi + (size_t)kB * kD;                      //  8 MB
  float* Zp = (float*)(ws + 2ull * kB * kD * sizeof(unsigned short));
  float* Pp = Zp + (size_t)SLOTS * kB;
  float* Np = Pp + (size_t)SLOTS * kB;

  const int nElem = kB * kD;
  infonce_split_kernel<<<(nElem + 255) / 256, 256, 0, stream>>>(desc, hi, lo, nElem);

  infonce_gram_kernel<<<(kB / ROWS) * COL_SPLIT, 256, 0, stream>>>(
      hi, lo, seq, tmid, Zp, Pp, Np);

  infonce_finalize_kernel<<<1, 256, 0, stream>>>(Zp, Pp, Np, out);
}
